// GATV2_Transformer_59468117180762
// MI455X (gfx1250) — matrix operand order repaired and checked by reading, not compiled, not run on hardware
//
#include <hip/hip_runtime.h>
#include <hip/hip_bf16.h>

// Problem constants (match reference)
#define N_NODES 2048
#define N_EDGES 32768
#define IN_F    256
#define DMODEL  128
#define HEADS   16
#define CDIM    128
#define HC      (HEADS * CDIM)   // 2048
#define FFDIM   2048

typedef __attribute__((ext_vector_type(16))) __bf16 v16bf;
typedef __attribute__((ext_vector_type(8)))  float  v8f;
typedef unsigned short ushort_t;

#ifndef __has_builtin
#define __has_builtin(x) 0
#endif
#if __has_builtin(__builtin_amdgcn_tensor_load_to_lds) && __has_builtin(__builtin_amdgcn_s_wait_tensorcnt)
#define USE_TDM 1
#else
#define USE_TDM 0
#endif

// ---------- helpers ----------
__device__ inline unsigned pk_bf16(float x, float y) {
  union { float f; unsigned u; } a, b;
  a.f = x; b.f = y;
  unsigned ra = (a.u + 0x7FFFu + ((a.u >> 16) & 1u)) >> 16;   // RNE f32 -> bf16
  unsigned rb = (b.u + 0x7FFFu + ((b.u >> 16) & 1u)) >> 16;
  return (ra & 0xFFFFu) | (rb << 16);
}

__device__ inline unsigned f2ord(float f) {   // monotone float -> uint (for atomicMax)
  union { float f; unsigned u; } x; x.f = f;
  return (x.u & 0x80000000u) ? ~x.u : (x.u | 0x80000000u);
}
__device__ inline float ord2f(unsigned o) {
  union { unsigned u; float f; } x;
  x.u = (o & 0x80000000u) ? (o ^ 0x80000000u) : ~o;
  return x.f;
}

#if USE_TDM
typedef unsigned int tdm_u32x4 __attribute__((ext_vector_type(4)));
typedef int          tdm_i32x4 __attribute__((ext_vector_type(4)));
typedef int          tdm_i32x8 __attribute__((ext_vector_type(8)));

// TDM: DMA a [32 rows(K) x 64 cols(N)] bf16 tile (row stride = ncols elements)
// from global memory into contiguous LDS (row-major, 128 B rows, no padding).
__device__ inline void tdm_load_b_tile(const ushort_t* gptr, int ncols, ushort_t* lds)
{
  const unsigned long long ga = (unsigned long long)gptr;
  tdm_u32x4 g0;
  g0.x = 1u;                                            // count=1, user descriptor
  g0.y = (unsigned)(unsigned long long)(uintptr_t)lds;  // LDS byte offset (addr[31:0])
  g0.z = (unsigned)(ga & 0xFFFFFFFFu);                  // global_addr[31:0]
  g0.w = (unsigned)((ga >> 32) & 0x1FFFFFFu) | (2u << 30); // addr[56:32] | type=2
  tdm_i32x8 g1;
  g1[0] = 1 << 16;      // workgroup_mask=0 (not in cluster), data_size=1 (2 bytes)
  g1[1] = 64 << 16;     // tensor_dim0 = 64 (== tile: no OOB)
  g1[2] = 32 << 16;     // tensor_dim0 hi=0, tensor_dim1 = 32
  g1[3] = 64 << 16;     // tensor_dim1 hi=0, tile_dim0 = 64
  g1[4] = 32;           // tile_dim1 = 32, tile_dim2 = 0 (2D)
  g1[5] = ncols;        // tensor_dim0_stride[31:0] (elements)
  g1[6] = 0;            // stride hi, tensor_dim1_stride lo (unused for 2D)
  g1[7] = 0;
  tdm_i32x4 z4 = {0, 0, 0, 0};
#if __clang_major__ >= 23
  tdm_i32x8 z8 = {0, 0, 0, 0, 0, 0, 0, 0};
  __builtin_amdgcn_tensor_load_to_lds(g0, g1, z4, z4, z8, 0);
#else
  __builtin_amdgcn_tensor_load_to_lds(g0, g1, z4, z4, 0);
#endif
}
#endif

// ---------- f32 -> packed bf16 conversion (vectorized, memory-bound) ----------
__global__ __launch_bounds__(256) void cvt_bf16_kernel(const float* __restrict__ in,
                                                       ushort_t* __restrict__ out, long n)
{
  const long i = (long)blockIdx.x * 1024 + (long)threadIdx.x * 4;
  if (i >= n) return;
  const float4 v = *(const float4*)(in + i);
  uint2 r;
  r.x = pk_bf16(v.x, v.y);
  r.y = pk_bf16(v.z, v.w);
  *(uint2*)(out + i) = r;
}

// ---------- WMMA bf16 GEMM: C[M,N] = act(A[M,K] @ B[K,N] + bias) ----------
// A, B pre-converted bf16, row-major. block = 128 threads = 4 waves.
// Block tile = 64(M) x 64(N); each wave computes 16x64 with 4 accumulators.
// B tile (32x64 per K-step) is staged into LDS by the Tensor Data Mover
// (wave 0 issues, double-buffered, overlapped with WMMA compute); A is read
// direct from global. grid = (N/64, M/64); K%32==0, N%64==0, M%64==0.
__global__ __launch_bounds__(128) void gemm_bf16_wmma(
    const ushort_t* __restrict__ A, const ushort_t* __restrict__ B,
    const float* __restrict__ bias, float* __restrict__ C,
    int M, int K, int Ncols, int relu)
{
  const int lane = threadIdx.x & 31;
  const int wv   = threadIdx.x >> 5;
  const int m0 = (blockIdx.y * 4 + wv) * 16;
  const int n0 = blockIdx.x * 64;

  const int kh = lane >> 4;                               // A: K-half held by this lane
  const ushort_t* pa_base = A + (long)(m0 + (lane & 15)) * K + kh * 8;

  v8f acc[4] = {{}, {}, {}, {}};
  union U { v16bf v; uint4 q[2]; };

#if USE_TDM
  __shared__ __align__(128) ushort_t Bs[2][32 * 64];      // double-buffered B tile (8 KB)
  const int nsteps = K >> 5;
  if (wv == 0) {
    tdm_load_b_tile(B + n0, Ncols, &Bs[0][0]);
    __builtin_amdgcn_s_wait_tensorcnt(0);
  }
  __syncthreads();
#pragma unroll 2
  for (int i = 0; i < nsteps; ++i) {
    const int k0 = i << 5;
    if (wv == 0 && i + 1 < nsteps)                        // prefetch next tile via TDM
      tdm_load_b_tile(B + (long)(k0 + 32) * Ncols + n0, Ncols, &Bs[(i + 1) & 1][0]);

    U a;
    const ushort_t* pa = pa_base + k0;
    __builtin_prefetch(pa + 32, 0, 0);
    a.q[0] = *(const uint4*)(pa);        // K = k0 + kh*8 + 0..7
    a.q[1] = *(const uint4*)(pa + 16);   // K = k0 + 16 + kh*8 + 0..7

    const ushort_t* bs = &Bs[i & 1][kh * 8 * 64 + (lane & 15)];     // B: lane holds col lane&15, k = k0 + kh*8 + {0..7, 16..23}
    U b[4];
#pragma unroll
    for (int t = 0; t < 4; ++t) {
      unsigned w[8];
#pragma unroll
      for (int j = 0; j < 8; ++j) {
        const int kk = (j < 4) ? 2 * j : 16 + 2 * (j - 4);
        w[j] = (unsigned)bs[kk * 64 + t * 16] | ((unsigned)bs[(kk + 1) * 64 + t * 16] << 16);
      }
      b[t].q[0] = make_uint4(w[0], w[1], w[2], w[3]);
      b[t].q[1] = make_uint4(w[4], w[5], w[6], w[7]);
    }
#pragma unroll
    for (int t = 0; t < 4; ++t)
      acc[t] = __builtin_amdgcn_wmma_f32_16x16x32_bf16(false, a.v, false, b[t].v,
                                                       (short)0, acc[t], false, false);
    if (wv == 0 && i + 1 < nsteps)
      __builtin_amdgcn_s_wait_tensorcnt(0);               // next tile landed in LDS
    __syncthreads();
  }
#else
  const ushort_t* pb_base = B + (long)(kh * 8) * Ncols + n0 + (lane & 15);
#pragma unroll 2
  for (int k0 = 0; k0 < K; k0 += 32) {
    U a;
    const ushort_t* pa = pa_base + k0;
    __builtin_prefetch(pa + 32, 0, 0);
    a.q[0] = *(const uint4*)(pa);
    a.q[1] = *(const uint4*)(pa + 16);
    const ushort_t* pb = pb_base + (long)k0 * Ncols;
    __builtin_prefetch(pb + 32 * Ncols, 0, 0);
    U b[4];
#pragma unroll
    for (int t = 0; t < 4; ++t) {
      unsigned w[8];
#pragma unroll
      for (int j = 0; j < 8; ++j) {
        const int kk = (j < 4) ? 2 * j : 16 + 2 * (j - 4);
        w[j] = (unsigned)pb[(long)kk * Ncols + t * 16] | ((unsigned)pb[(long)(kk + 1) * Ncols + t * 16] << 16);
      }
      b[t].q[0] = make_uint4(w[0], w[1], w[2], w[3]);
      b[t].q[1] = make_uint4(w[4], w[5], w[6], w[7]);
    }
#pragma unroll
    for (int t = 0; t < 4; ++t)
      acc[t] = __builtin_amdgcn_wmma_f32_16x16x32_bf16(false, a.v, false, b[t].v,
                                                       (short)0, acc[t], false, false);
  }
#endif

  // C/D layout: vgpr r -> row m0 + r + 8*(lane>>4), col = (lane&15) within subtile
  const int col   = lane & 15;
  const int rbase = m0 + ((lane >> 4) << 3);
#pragma unroll
  for (int t = 0; t < 4; ++t) {
    const int ccol = n0 + t * 16 + col;
    const float bv = bias ? bias[ccol] : 0.0f;
#pragma unroll
    for (int r = 0; r < 8; ++r) {
      float v = acc[t][r] + bv;
      if (relu) v = fmaxf(v, 0.f);
      C[(long)(rbase + r) * Ncols + ccol] = v;
    }
  }
}

// ---------- GAT edge kernels ----------
__global__ __launch_bounds__(256) void gat_logits_kernel(
    const float* __restrict__ xl, const float* __restrict__ xr,
    const float* __restrict__ ea, const int* __restrict__ src,
    const int* __restrict__ dst, const float* __restrict__ we,
    const float* __restrict__ att, float* __restrict__ logits,
    unsigned* __restrict__ mxb)
{
  const int idx = blockIdx.x * 256 + threadIdx.x;
  const int e = idx >> 4, h = idx & 15;
  const int is = src[e], id = dst[e];
  const float a = ea[e];
  const float4* pl = (const float4*)(xl + (long)is * HC + h * CDIM);
  const float4* pr = (const float4*)(xr + (long)id * HC + h * CDIM);
  const float4* pw = (const float4*)(we + h * CDIM);
  const float4* pt = (const float4*)(att + h * CDIM);
  float s = 0.f;
#pragma unroll 4
  for (int c = 0; c < CDIM / 4; ++c) {
    float4 vl = pl[c], vr = pr[c], vw = pw[c], vt = pt[c];
    float m0 = vl.x + vr.x + a * vw.x; m0 = m0 > 0.f ? m0 : 0.2f * m0;
    float m1 = vl.y + vr.y + a * vw.y; m1 = m1 > 0.f ? m1 : 0.2f * m1;
    float m2 = vl.z + vr.z + a * vw.z; m2 = m2 > 0.f ? m2 : 0.2f * m2;
    float m3 = vl.w + vr.w + a * vw.w; m3 = m3 > 0.f ? m3 : 0.2f * m3;
    s += m0 * vt.x + m1 * vt.y + m2 * vt.z + m3 * vt.w;
  }
  logits[idx] = s;
  atomicMax(&mxb[(long)id * HEADS + h], f2ord(s));
}

__global__ __launch_bounds__(256) void gat_exp_kernel(
    const float* __restrict__ logits, const int* __restrict__ dst,
    const unsigned* __restrict__ mxb, float* __restrict__ ex,
    float* __restrict__ den)
{
  const int idx = blockIdx.x * 256 + threadIdx.x;
  const int e = idx >> 4, h = idx & 15;
  const int id = dst[e];
  const float mxv = ord2f(mxb[(long)id * HEADS + h]);
  const float ev = __expf(logits[idx] - mxv);
  ex[idx] = ev;
  atomicAdd(&den[(long)id * HEADS + h], ev);
}

__global__ __launch_bounds__(256) void g_init_kernel(float* __restrict__ g,
                                                     const float* __restrict__ bias)
{
  const int i = blockIdx.x * 256 + threadIdx.x;
  g[i] = bias[i & (HC - 1)];
}

__global__ __launch_bounds__(256) void g_accum_kernel(
    const float* __restrict__ xl, const int* __restrict__ src,
    const int* __restrict__ dst, const float* __restrict__ ex,
    const float* __restrict__ den, float* __restrict__ g)
{
  const int idx = blockIdx.x * 256 + threadIdx.x;
  const int e = idx >> 4, h = idx & 15;
  const int is = src[e], id = dst[e];
  const float alpha = ex[idx] / den[(long)id * HEADS + h];
  const float4* pl = (const float4*)(xl + (long)is * HC + h * CDIM);
  float* pg = g + (long)id * HC + h * CDIM;
#pragma unroll 2
  for (int c = 0; c < CDIM / 4; ++c) {
    float4 v = pl[c];
    atomicAdd(&pg[4 * c + 0], alpha * v.x);
    atomicAdd(&pg[4 * c + 1], alpha * v.y);
    atomicAdd(&pg[4 * c + 2], alpha * v.z);
    atomicAdd(&pg[4 * c + 3], alpha * v.w);
  }
}

// ---------- attention: flash-style, DH=8, one thread per (head,row) ----------
__global__ __launch_bounds__(256) void attn_kernel(const float* __restrict__ qkv,
                                                   float* __restrict__ o)
{
  const int h = blockIdx.y;
  const int n = blockIdx.x * 256 + threadIdx.x;
  __shared__ float Ks[256 * 8];
  __shared__ float Vs[256 * 8];
  float q[8];
#pragma unroll
  for (int d = 0; d < 8; ++d) q[d] = qkv[(long)n * 384 + h * 8 + d];
  float mrun = -3.4e38f, lrun = 0.f, acc[8] = {};
  const float scale = 0.35355339059327379f; // 1/sqrt(8)
  for (int c0 = 0; c0 < N_NODES; c0 += 256) {
    __syncthreads();
    const int r = c0 + threadIdx.x;
#pragma unroll
    for (int d = 0; d < 8; ++d) {
      Ks[threadIdx.x * 8 + d] = qkv[(long)r * 384 + 128 + h * 8 + d];
      Vs[threadIdx.x * 8 + d] = qkv[(long)r * 384 + 256 + h * 8 + d];
    }
    __syncthreads();
    for (int m = 0; m < 256; ++m) {
      float s = 0.f;
#pragma unroll
      for (int d = 0; d < 8; ++d) s += q[d] * Ks[m * 8 + d];
      s *= scale;
      const float mnew = fmaxf(mrun, s);
      const float corr = __expf(mrun - mnew);
      const float p    = __expf(s - mnew);
      lrun = lrun * corr + p;
#pragma unroll
      for (int d = 0; d < 8; ++d) acc[d] = acc[d] * corr + p * Vs[m * 8 + d];
      mrun = mnew;
    }
  }
  const float inv = 1.f / lrun;
#pragma unroll
  for (int d = 0; d < 8; ++d) o[(long)n * DMODEL + h * 8 + d] = acc[d] * inv;
}

// ---------- layernorm(residual) ----------
__global__ __launch_bounds__(128) void ln_residual(
    const float* __restrict__ a, const float* __restrict__ r,
    const float* __restrict__ gam, const float* __restrict__ bet,
    float* __restrict__ out)
{
  __shared__ float sd[128];
  const int row = blockIdx.x, tid = threadIdx.x;
  const float v = a[row * DMODEL + tid] + r[row * DMODEL + tid];
  sd[tid] = v; __syncthreads();
  for (int s = 64; s > 0; s >>= 1) { if (tid < s) sd[tid] += sd[tid + s]; __syncthreads(); }
  const float mu = sd[0] * (1.f / DMODEL); __syncthreads();
  const float d = v - mu;
  sd[tid] = d * d; __syncthreads();
  for (int s = 64; s > 0; s >>= 1) { if (tid < s) sd[tid] += sd[tid + s]; __syncthreads(); }
  const float var = sd[0] * (1.f / DMODEL);
  out[row * DMODEL + tid] = d * rsqrtf(var + 1e-5f) * gam[tid] + bet[tid];
}

// ---------- misc elementwise ----------
__global__ __launch_bounds__(256) void fill_u32(unsigned* __restrict__ p, unsigned v, int n)
{
  const int i = blockIdx.x * 256 + threadIdx.x;
  if (i < n) p[i] = v;
}

__global__ __launch_bounds__(256) void fuse_kernel(const float* __restrict__ t2,
                                                   const float* __restrict__ glp,
                                                   float* __restrict__ ebd)
{
  const int i = blockIdx.x * 256 + threadIdx.x;
  const float s = 1.f / (1.f + __expf(-t2[i]));
  ebd[i] = s * glp[i];
}

// out[N,2] = Hc[N,2048] @ W[2048,2] + b  (N=2 too small for WMMA tiles)
__global__ __launch_bounds__(256) void cls2_kernel(const float* __restrict__ Hc,
                                                   const float* __restrict__ W,
                                                   const float* __restrict__ b,
                                                   float* __restrict__ out)
{
  const int idx = blockIdx.x * 256 + threadIdx.x;
  const int nrow = idx >> 1, j = idx & 1;
  float s = b[j];
  const float* hp = Hc + (long)nrow * FFDIM;
  for (int k = 0; k < FFDIM; ++k) s += hp[k] * W[k * 2 + j];
  out[idx] = s;
}

// ---------- driver ----------
static inline void cvt(const float* in, ushort_t* out, long n, hipStream_t s) {
  cvt_bf16_kernel<<<dim3((unsigned)((n + 1023) / 1024)), dim3(256), 0, s>>>(in, out, n);
}

extern "C" void kernel_launch(void* const* d_in, const int* in_sizes, int n_in,
                              void* d_out, int out_size, void* d_ws, size_t ws_size,
                              hipStream_t stream) {
  const float* x        = (const float*)d_in[0];
  const int*   eidx     = (const int*)d_in[1];
  const float* eattr    = (const float*)d_in[2];
  const float* enc_w1   = (const float*)d_in[3];
  const float* enc_b1   = (const float*)d_in[4];
  const float* enc_w2   = (const float*)d_in[5];
  const float* enc_b2   = (const float*)d_in[6];
  const float* gat_wl   = (const float*)d_in[7];
  const float* gat_bl   = (const float*)d_in[8];
  const float* gat_wr   = (const float*)d_in[9];
  const float* gat_br   = (const float*)d_in[10];
  const float* gat_we   = (const float*)d_in[11];
  const float* gat_att  = (const float*)d_in[12];
  const float* gat_bias = (const float*)d_in[13];
  const float* ipw      = (const float*)d_in[14];
  const float* ipb      = (const float*)d_in[15];
  const float* opw      = (const float*)d_in[16];
  const float* opb      = (const float*)d_in[17];
  const float* ffw1     = (const float*)d_in[18];
  const float* ffb1     = (const float*)d_in[19];
  const float* ffw2     = (const float*)d_in[20];
  const float* ffb2     = (const float*)d_in[21];
  const float* ln1g     = (const float*)d_in[22];
  const float* ln1b     = (const float*)d_in[23];
  const float* ln2g     = (const float*)d_in[24];
  const float* ln2b     = (const float*)d_in[25];
  const float* glw      = (const float*)d_in[26];
  const float* glb      = (const float*)d_in[27];
  const float* cw1      = (const float*)d_in[28];
  const float* cb1      = (const float*)d_in[29];
  const float* cw2      = (const float*)d_in[30];
  const float* cb2      = (const float*)d_in[31];

  // ---- workspace layout: f32 regions (aliased across stages), then bf16 staging
  float* ws     = (float*)d_ws;
  float* bigA   = ws;                        // 4M f32 : enc_h -> xl -> cls_h
  float* bigB   = bigA + 4194304;            // 4M f32 : xr -> ff_h
  float* g      = bigB + 4194304;            // 4M f32
  float* enc    = g + 4194304;               // 256K
  float* qkv    = enc + 262144;              // 768K
  float* logits = qkv + 786432;              // 512K
  float* ex     = logits + 524288;           // 512K
  unsigned* mxb = (unsigned*)(ex + 524288);  // 32K uint keys
  float* den    = (float*)(mxb + 32768);     // 32K
  float* ao     = den + 32768;               // 256K
  float* tmp    = ao + 262144;               // 256K
  float* t1     = tmp + 262144;              // 256K
  float* t2     = t1 + 262144;               // 256K
  float* glp    = t2 + 262144;               // 256K
  float* ebd    = glp + 262144;              // 256K
  // bf16 staging (ushort elements), starts right after f32 part
  ushort_t* w1b   = (ushort_t*)(ebd + 262144);
  ushort_t* w2b   = w1b + 131072;
  ushort_t* wlb   = w2b + 65536;
  ushort_t* wrb   = wlb + 262144;
  ushort_t* ipwb  = wrb + 262144;
  ushort_t* opwb  = ipwb + 49152;
  ushort_t* ffw1b = opwb + 16384;
  ushort_t* ffw2b = ffw1b + 262144;
  ushort_t* glwb  = ffw2b + 262144;
  ushort_t* cw1b  = glwb + 262144;
  ushort_t* xbf   = cw1b + 262144;   // 524288
  ushort_t* encbf = xbf + 524288;    // 262144 (enc reused by 3 GEMMs)
  ushort_t* abf   = encbf + 262144;  // 4194304 generic activation staging

  const int* esrc = eidx;
  const int* edst = eidx + N_EDGES;

  const dim3 blk128(128), blk256(256);
  const int MY = N_NODES / 64;   // gemm grid.y

  // ---- convert weights + input once (memory-bound, amortized) ----
  cvt(x, xbf, (long)N_NODES * IN_F, stream);
  cvt(enc_w1, w1b, (long)IN_F * 512, stream);
  cvt(enc_w2, w2b, (long)512 * DMODEL, stream);
  cvt(gat_wl, wlb, (long)DMODEL * HC, stream);
  cvt(gat_wr, wrb, (long)DMODEL * HC, stream);
  cvt(ipw, ipwb, (long)DMODEL * 384, stream);
  cvt(opw, opwb, (long)DMODEL * DMODEL, stream);
  cvt(ffw1, ffw1b, (long)DMODEL * FFDIM, stream);
  cvt(ffw2, ffw2b, (long)FFDIM * DMODEL, stream);
  cvt(glw, glwb, (long)HC * DMODEL, stream);
  cvt(cw1, cw1b, (long)DMODEL * FFDIM, stream);

  // ---- node encoder ----
  gemm_bf16_wmma<<<dim3(512 / 64, MY), blk128, 0, stream>>>(xbf, w1b, enc_b1, bigA, N_NODES, IN_F, 512, 1);
  cvt(bigA, abf, (long)N_NODES * 512, stream);
  gemm_bf16_wmma<<<dim3(DMODEL / 64, MY), blk128, 0, stream>>>(abf, w2b, enc_b2, enc, N_NODES, 512, DMODEL, 0);
  cvt(enc, encbf, (long)N_NODES * DMODEL, stream);

  // ---- GAT projections ----
  gemm_bf16_wmma<<<dim3(HC / 64, MY), blk128, 0, stream>>>(encbf, wlb, gat_bl, bigA, N_NODES, DMODEL, HC, 0);
  gemm_bf16_wmma<<<dim3(HC / 64, MY), blk128, 0, stream>>>(encbf, wrb, gat_br, bigB, N_NODES, DMODEL, HC, 0);

  // ---- edge softmax + scatter (mxb and den contiguous: zero both in one pass) ----
  fill_u32<<<65536 / 256, blk256, 0, stream>>>(mxb, 0u, 65536);
  gat_logits_kernel<<<(N_EDGES * HEADS) / 256, blk256, 0, stream>>>(bigA, bigB, eattr, esrc, edst, gat_we, gat_att, logits, mxb);
  gat_exp_kernel<<<(N_EDGES * HEADS) / 256, blk256, 0, stream>>>(logits, edst, mxb, ex, den);
  g_init_kernel<<<(N_NODES * HC) / 256, blk256, 0, stream>>>(g, gat_bias);
  g_accum_kernel<<<(N_EDGES * HEADS) / 256, blk256, 0, stream>>>(bigA, esrc, edst, ex, den, g);

  // ---- transformer layer ----
  gemm_bf16_wmma<<<dim3(384 / 64, MY), blk128, 0, stream>>>(encbf, ipwb, ipb, qkv, N_NODES, DMODEL, 384, 0);
  attn_kernel<<<dim3(N_NODES / 256, HEADS), blk256, 0, stream>>>(qkv, ao);
  cvt(ao, abf, (long)N_NODES * DMODEL, stream);
  gemm_bf16_wmma<<<dim3(DMODEL / 64, MY), blk128, 0, stream>>>(abf, opwb, opb, tmp, N_NODES, DMODEL, DMODEL, 0);
  ln_residual<<<N_NODES, blk128, 0, stream>>>(enc, tmp, ln1g, ln1b, t1);
  cvt(t1, abf, (long)N_NODES * DMODEL, stream);
  gemm_bf16_wmma<<<dim3(FFDIM / 64, MY), blk128, 0, stream>>>(abf, ffw1b, ffb1, bigB, N_NODES, DMODEL, FFDIM, 1);
  cvt(bigB, abf, (long)N_NODES * FFDIM, stream);
  gemm_bf16_wmma<<<dim3(DMODEL / 64, MY), blk128, 0, stream>>>(abf, ffw2b, ffb2, tmp, N_NODES, FFDIM, DMODEL, 0);
  ln_residual<<<N_NODES, blk128, 0, stream>>>(t1, tmp, ln2g, ln2b, t2);

  // ---- fuse + classify ----
  cvt(g, abf, (long)N_NODES * HC, stream);
  gemm_bf16_wmma<<<dim3(DMODEL / 64, MY), blk128, 0, stream>>>(abf, glwb, glb, glp, N_NODES, HC, DMODEL, 0);
  fuse_kernel<<<(N_NODES * DMODEL) / 256, blk256, 0, stream>>>(t2, glp, ebd);
  cvt(ebd, abf, (long)N_NODES * DMODEL, stream);
  gemm_bf16_wmma<<<dim3(FFDIM / 64, MY), blk128, 0, stream>>>(abf, cw1b, cb1, bigA, N_NODES, DMODEL, FFDIM, 1);
  cls2_kernel<<<(N_NODES * 2) / 256, blk256, 0, stream>>>(bigA, cw2, cb2, (float*)d_out);
}
